// TpnTaskLoss_54743653155312
// MI455X (gfx1250) — matrix operand order repaired and checked by reading, not compiled, not run on hardware
//
#include <hip/hip_runtime.h>
#include <hip/hip_bf16.h>
#include <math.h>

// ---------------------------------------------------------------------------
// TpnTaskLoss for MI455X (gfx1250, wave32, WMMA).
// 3 GEMMs [32768x512]@[512x256] in bf16 WMMA (f32 accum), fused with
// log-softmax + 6-way KL reduction. M-tile 32 rows/block, each B fragment
// feeds 2 WMMAs; all loads of a k-step hoisted ahead of the WMMA burst.
// ---------------------------------------------------------------------------

typedef __bf16 bf16_t;
typedef __attribute__((ext_vector_type(16))) __bf16 v16bf;
typedef __attribute__((ext_vector_type(8)))  __bf16 v8bf;
typedef __attribute__((ext_vector_type(8)))  float  v8f;

#define NSRC 16384
#define DDIM 512
#define NCLS 256
#define MROWS (2 * NSRC)

// ---------------------------------------------------------------------------
__global__ void zero_kernel(float* __restrict__ p, int n) {
    int i = blockIdx.x * 256 + threadIdx.x;
    if (i < n) p[i] = 0.0f;
}

// ---------------------------------------------------------------------------
// segment sums + counts (global f32 atomics) + f32 -> bf16 of features
// ---------------------------------------------------------------------------
__global__ void seg_kernel(const float* __restrict__ src, const float* __restrict__ trg,
                           const int* __restrict__ slab, const int* __restrict__ tlab,
                           float* __restrict__ sum_s, float* __restrict__ sum_t,
                           float* __restrict__ cnt_s, float* __restrict__ cnt_t,
                           bf16_t* __restrict__ Fbf) {
    int row = blockIdx.x;                     // 0 .. MROWS-1
    bool is_src = row < NSRC;
    const float* f = is_src ? (src + (size_t)row * DDIM)
                            : (trg + (size_t)(row - NSRC) * DDIM);
    int lab = is_src ? slab[row] : tlab[row - NSRC];
    float* sums = is_src ? sum_s : sum_t;
    for (int d = threadIdx.x; d < DDIM; d += 256) {
        float v = f[d];
        Fbf[(size_t)row * DDIM + d] = (bf16_t)v;
        atomicAdd(&sums[(size_t)lab * DDIM + d], v);
    }
    if (threadIdx.x == 0)
        atomicAdd(is_src ? &cnt_s[lab] : &cnt_t[lab], 1.0f);
}

// ---------------------------------------------------------------------------
// class means u_s, u_t, u_st -> bf16 [3*256 x 512] (row c == B column c)
// ---------------------------------------------------------------------------
__global__ void mean_kernel(const float* __restrict__ sum_s, const float* __restrict__ sum_t,
                            const float* __restrict__ cnt_s, const float* __restrict__ cnt_t,
                            bf16_t* __restrict__ U) {
    int c = blockIdx.x;
    float ns = cnt_s[c], nt = cnt_t[c];
    float rs = 1.0f / ns, rt = 1.0f / nt, rst = 1.0f / (ns + nt);
    for (int d = threadIdx.x; d < DDIM; d += 256) {
        float ss = sum_s[(size_t)c * DDIM + d];
        float st = sum_t[(size_t)c * DDIM + d];
        U[((size_t)(0 * NCLS + c)) * DDIM + d] = (bf16_t)(ss * rs);
        U[((size_t)(1 * NCLS + c)) * DDIM + d] = (bf16_t)(st * rt);
        U[((size_t)(2 * NCLS + c)) * DDIM + d] = (bf16_t)((ss + st) * rst);
    }
}

// ---------------------------------------------------------------------------
// fused GEMM (bf16 WMMA) + log-softmax + 6-way KL accumulation.
// Block = 32-row tile (2 slabs of 16), 8 waves; wave w owns column tiles
// {2w, 2w+1} of each of the 3 mean matrices, for both slabs: 12 acc tiles.
// ---------------------------------------------------------------------------
__global__ __launch_bounds__(256, 1)
void gemm_kl_kernel(const bf16_t* __restrict__ F, const bf16_t* __restrict__ U,
                    float* __restrict__ total) {
    __shared__ float zl[3][16][NCLS];   // logits for one 16-row slab (48KB)
    __shared__ float block_sum;

    const int tid  = threadIdx.x;
    const int w    = tid >> 5;          // wave 0..7
    const int lane = tid & 31;
    const int half = lane >> 4;         // 0: lanes 0-15, 1: lanes 16-31
    const int r    = lane & 15;
    const int m0   = blockIdx.x * 32;

    v8f acc[2][3][2];                   // [slab][matrix][coltile]
#pragma unroll
    for (int s = 0; s < 2; ++s)
#pragma unroll
        for (int mi = 0; mi < 3; ++mi)
#pragma unroll
            for (int ti = 0; ti < 2; ++ti)
                acc[s][mi][ti] = (v8f){0.f, 0.f, 0.f, 0.f, 0.f, 0.f, 0.f, 0.f};

    const bf16_t* arow0 = F + (size_t)(m0 + r) * DDIM;
    const bf16_t* arow1 = arow0 + (size_t)16 * DDIM;
    const bf16_t* brow[3][2];
#pragma unroll
    for (int mi = 0; mi < 3; ++mi)
#pragma unroll
        for (int ti = 0; ti < 2; ++ti)
            brow[mi][ti] = U + (size_t)(mi * NCLS + 32 * w + 16 * ti + r) * DDIM;

    for (int kb = 0; kb < DDIM; kb += 32) {
        // ---- load phase: everything this k-step needs, before any WMMA ----
        // A fragments (16x32 bf16, ISA layout): lanes 0-15 hold K {0..7,16..23},
        // lanes 16-31 hold K {8..15,24..31} of the chunk.
        v8bf a0l = *(const v8bf*)(arow0 + kb + half * 8);
        v8bf a0h = *(const v8bf*)(arow0 + kb + 16 + half * 8);
        v8bf a1l = *(const v8bf*)(arow1 + kb + half * 8);
        v8bf a1h = *(const v8bf*)(arow1 + kb + 16 + half * 8);
        // B fragments (32x16 bf16): lane n holds B column n = U row n;
        // lanes 0-15: K {0..7,16..23}, lanes 16-31: K {8..15,24..31}.
        v8bf bl[3][2], bh[3][2];
#pragma unroll
        for (int mi = 0; mi < 3; ++mi)
#pragma unroll
            for (int ti = 0; ti < 2; ++ti) {
                const bf16_t* bp = brow[mi][ti] + kb + half * 8;
                bl[mi][ti] = *(const v8bf*)(bp);
                bh[mi][ti] = *(const v8bf*)(bp + 16);
            }

        v16bf af0, af1;
#pragma unroll
        for (int i = 0; i < 8; ++i) {
            af0[i] = a0l[i]; af0[i + 8] = a0h[i];
            af1[i] = a1l[i]; af1[i + 8] = a1h[i];
        }

        // ---- compute phase: 12 WMMAs back-to-back, each B used twice ----
#pragma unroll
        for (int mi = 0; mi < 3; ++mi)
#pragma unroll
            for (int ti = 0; ti < 2; ++ti) {
                v16bf bf;
#pragma unroll
                for (int i = 0; i < 8; ++i) {
                    bf[i] = bl[mi][ti][i]; bf[i + 8] = bh[mi][ti][i];
                }
                acc[0][mi][ti] = __builtin_amdgcn_wmma_f32_16x16x32_bf16(
                    false, af0, false, bf, (short)0, acc[0][mi][ti], false, false);
                acc[1][mi][ti] = __builtin_amdgcn_wmma_f32_16x16x32_bf16(
                    false, af1, false, bf, (short)0, acc[1][mi][ti], false, false);
            }
    }

    if (tid == 0) block_sum = 0.0f;

    // ---- epilogue per slab: C layout VGPR v, lanes 0-15 -> (M=v, N=lane);
    //      lanes 16-31 -> (M=v+8, N=lane-16). ----
    for (int slab = 0; slab < 2; ++slab) {
        __syncthreads();   // slab0: after block_sum init; slab1: after prior reads
#pragma unroll
        for (int mi = 0; mi < 3; ++mi)
#pragma unroll
            for (int ti = 0; ti < 2; ++ti) {
                int col = 32 * w + 16 * ti + r;
#pragma unroll
                for (int v = 0; v < 8; ++v)
                    zl[mi][v + 8 * half][col] = acc[slab][mi][ti][v];
            }
        __syncthreads();

        // wave w handles rows {w, w+8}; lane covers cols lane+32j, j=0..7
        for (int rr = w; rr < 16; rr += 8) {
            float mx0 = -INFINITY, mx1 = -INFINITY, mx2 = -INFINITY;
#pragma unroll
            for (int j = 0; j < 8; ++j) {
                int c = lane + 32 * j;
                mx0 = fmaxf(mx0, zl[0][rr][c]);
                mx1 = fmaxf(mx1, zl[1][rr][c]);
                mx2 = fmaxf(mx2, zl[2][rr][c]);
            }
#pragma unroll
            for (int off = 16; off > 0; off >>= 1) {
                mx0 = fmaxf(mx0, __shfl_xor(mx0, off, 32));
                mx1 = fmaxf(mx1, __shfl_xor(mx1, off, 32));
                mx2 = fmaxf(mx2, __shfl_xor(mx2, off, 32));
            }
            float se0 = 0.f, se1 = 0.f, se2 = 0.f;
#pragma unroll
            for (int j = 0; j < 8; ++j) {
                int c = lane + 32 * j;
                se0 += __expf(zl[0][rr][c] - mx0);
                se1 += __expf(zl[1][rr][c] - mx1);
                se2 += __expf(zl[2][rr][c] - mx2);
            }
#pragma unroll
            for (int off = 16; off > 0; off >>= 1) {
                se0 += __shfl_xor(se0, off, 32);
                se1 += __shfl_xor(se1, off, 32);
                se2 += __shfl_xor(se2, off, 32);
            }
            float lg0 = mx0 + __logf(se0);
            float lg1 = mx1 + __logf(se1);
            float lg2 = mx2 + __logf(se2);

            float part = 0.0f;
#pragma unroll
            for (int j = 0; j < 8; ++j) {
                int c = lane + 32 * j;
                float ls  = zl[0][rr][c] - lg0;
                float lt  = zl[1][rr][c] - lg1;
                float lst = zl[2][rr][c] - lg2;
                float ps  = __expf(ls);
                float pt  = __expf(lt);
                float pst = __expf(lst);
                part += ps  * (2.0f * ls  - lt - lst)
                      + pt  * (2.0f * lt  - ls - lst)
                      + pst * (2.0f * lst - ls - lt);
            }
#pragma unroll
            for (int off = 16; off > 0; off >>= 1)
                part += __shfl_xor(part, off, 32);
            if (lane == 0) atomicAdd(&block_sum, part);
        }
    }
    __syncthreads();
    if (tid == 0) atomicAdd(total, block_sum);
}

// ---------------------------------------------------------------------------
__global__ void fin_kernel(const float* __restrict__ total, float* __restrict__ out) {
    out[0] = total[0] * (1.0f / (6.0f * (float)MROWS * (float)NCLS));
}

// ---------------------------------------------------------------------------
extern "C" void kernel_launch(void* const* d_in, const int* in_sizes, int n_in,
                              void* d_out, int out_size, void* d_ws, size_t ws_size,
                              hipStream_t stream) {
    const float* src_feat = (const float*)d_in[0];
    const float* trg_feat = (const float*)d_in[1];
    const int*   src_lab  = (const int*)d_in[2];
    const int*   trg_lab  = (const int*)d_in[3];
    float* out = (float*)d_out;

    float* sum_s = (float*)d_ws;                    // 256*512
    float* sum_t = sum_s + NCLS * DDIM;             // 256*512
    float* cnt_s = sum_t + NCLS * DDIM;             // 256
    float* cnt_t = cnt_s + NCLS;                    // 256
    float* total = cnt_t + NCLS;                    // 1 (+15 pad)
    bf16_t* Ubf  = (bf16_t*)(total + 16);           // 3*256*512 bf16 (B operand)
    bf16_t* Fbf  = Ubf + (size_t)3 * NCLS * DDIM;   // 32768*512 bf16 (A operand)

    const int nzero = 2 * NCLS * DDIM + 2 * NCLS + 16;
    zero_kernel<<<(nzero + 255) / 256, 256, 0, stream>>>(sum_s, nzero);

    seg_kernel<<<MROWS, 256, 0, stream>>>(src_feat, trg_feat, src_lab, trg_lab,
                                          sum_s, sum_t, cnt_s, cnt_t, Fbf);

    mean_kernel<<<NCLS, 256, 0, stream>>>(sum_s, sum_t, cnt_s, cnt_t, Ubf);

    gemm_kl_kernel<<<MROWS / 32, 256, 0, stream>>>(Fbf, Ubf, total);

    fin_kernel<<<1, 32, 0, stream>>>(total, out);
}
